// LocalAttention_24206435680836
// MI455X (gfx1250) — matrix operand order repaired and checked by reading, compiled, not run on hardware
//
#include <hip/hip_runtime.h>
#include <hip/hip_bf16.h>

typedef __attribute__((ext_vector_type(16))) _Float16 v16h;
typedef __attribute__((ext_vector_type(8)))  float    v8f;

#define T_DIM 4096
#define B_DIM 32
#define H_DIM 256
#define L_DIM 512
#define K_DIM 32

// ---------------------------------------------------------------------------
// Kernel 1: kernel = (lm_state @ Wk.T + bk)  -> f16, shape (B, H*K) flat.
// M=32 (2 tiles), N=8192 (512 tiles), K=512 (16 steps of 32).
// One 16x16 output tile per wave; 1024 waves = 128 blocks x 8 waves.
// ---------------------------------------------------------------------------
__global__ void wk_gemm_kernel(const float* __restrict__ lm,
                               const float* __restrict__ Wk,
                               const float* __restrict__ bk,
                               _Float16* __restrict__ kern_out) {
    const int wid   = blockIdx.x * 8 + (threadIdx.x >> 5);
    const int lane  = threadIdx.x & 31;
    const int g     = lane >> 4;     // lane half (K-group select)
    const int mr    = lane & 15;
    const int ntile = wid & 511;
    const int mtile = wid >> 9;
    const int m     = mtile * 16 + mr;   // A row
    const int n     = ntile * 16 + mr;   // B col / D col

    v8f c = {};
    for (int ks = 0; ks < 16; ++ks) {
        const int kb = ks * 32;
        v16h a, bf;
#pragma unroll
        for (int i = 0; i < 16; ++i) {
            // A-matrix 16-bit layout: k = i + 8*g + (i>=8 ? 8 : 0)
            const int ka = kb + i + 8 * g + ((i >= 8) ? 8 : 0);
            a[i] = (_Float16)lm[m * L_DIM + ka];
            // B-matrix 16-bit layout: k = i + 8*g + (i>=8 ? 8 : 0)  (B[k][n] = Wk[n][k])
            const int kw = kb + i + 8 * g + ((i >= 8) ? 8 : 0);
            bf[i] = (_Float16)Wk[n * L_DIM + kw];
        }
        c = __builtin_amdgcn_wmma_f32_16x16x32_f16(false, a, false, bf,
                                                   (short)0, c, false, false);
    }
    const float bias = bk[n];
#pragma unroll
    for (int r = 0; r < 8; ++r) {
        const int mo = mtile * 16 + r + 8 * g;           // D row
        kern_out[mo * (H_DIM * K_DIM) + n] = (_Float16)(c[r] + bias);
    }
}

// ---------------------------------------------------------------------------
// Kernel 2: global_hidden = lm_state @ Wg.T + bg  (B,H) f32 — tiny, scalar.
// ---------------------------------------------------------------------------
__global__ void ghid_kernel(const float* __restrict__ lm,
                            const float* __restrict__ Wg,
                            const float* __restrict__ bg,
                            float* __restrict__ ghid) {
    const int idx = blockIdx.x * blockDim.x + threadIdx.x;   // 0..8191
    const int b = idx >> 8;
    const int h = idx & 255;
    float acc = bg[h];
    for (int l = 0; l < L_DIM; ++l)
        acc = fmaf(lm[b * L_DIM + l], Wg[h * L_DIM + l], acc);
    ghid[idx] = acc;
}

// ---------------------------------------------------------------------------
// Kernel 3 (fused, streams encoded once):
//   per (b, 16-row T tile per wave): A = windows (16x32 f16, built once),
//   loop 16 H-tiles: B = kernel_b^T (32x16 f16), C = encoded + global (f32),
//   D = WMMA(A,B,C) = hidden;  partial[t] += Ws[h]*tanh(D);  xor-reduce.
// ---------------------------------------------------------------------------
__global__ void fused_scores_kernel(const float* __restrict__ enc,
                                    const float* __restrict__ maskp,
                                    const float* __restrict__ prev,
                                    const _Float16* __restrict__ kern,
                                    const float* __restrict__ ghid,
                                    const float* __restrict__ Ws,
                                    const float* __restrict__ bs,
                                    float* __restrict__ scores) {
    const int b     = blockIdx.x & 31;
    const int ttile = blockIdx.x >> 5;
    const int wave  = threadIdx.x >> 5;
    const int lane  = threadIdx.x & 31;
    const int g     = lane >> 4;
    const int mr    = lane & 15;
    const int t0    = ttile * 128 + wave * 16;

    // Build A fragment: windows[t, k] = prev[t + k - (K-1), b]  (zero-padded)
    const int trow = t0 + mr;
    v16h a;
#pragma unroll
    for (int i = 0; i < 16; ++i) {
        const int k  = i + 16 * g;
        const int st = trow + k - (K_DIM - 1);
        a[i] = (st >= 0) ? (_Float16)prev[st * B_DIM + b] : (_Float16)0.0f;
    }

    float partial[8];
#pragma unroll
    for (int r = 0; r < 8; ++r) partial[r] = 0.0f;

    const _Float16* kb = kern + b * (H_DIM * K_DIM);
    for (int ht = 0; ht < 16; ++ht) {
        const int h = ht * 16 + mr;                // column (N) for this lane
        v16h bf;
#pragma unroll
        for (int i = 0; i < 16; ++i)               // 16 consecutive halves
            bf[i] = kb[h * K_DIM + 16 * g + i];

        const float gh = ghid[b * H_DIM + h];
        v8f c;
#pragma unroll
        for (int r = 0; r < 8; ++r) {
            const int t = t0 + r + 8 * g;
            c[r] = enc[(t * B_DIM + b) * H_DIM + h] + gh;
        }
        c = __builtin_amdgcn_wmma_f32_16x16x32_f16(false, a, false, bf,
                                                   (short)0, c, false, false);
        const float w = Ws[h];
#pragma unroll
        for (int r = 0; r < 8; ++r)
            partial[r] += w * tanhf(c[r]);
    }

    // Reduce over the 16 lanes of each half-wave (N dimension of the tile).
#pragma unroll
    for (int r = 0; r < 8; ++r) {
        partial[r] += __shfl_xor(partial[r], 1);
        partial[r] += __shfl_xor(partial[r], 2);
        partial[r] += __shfl_xor(partial[r], 4);
        partial[r] += __shfl_xor(partial[r], 8);
    }
    if (mr < 8) {
        const int t = t0 + mr + 8 * g;
        scores[t * B_DIM + b] = partial[mr] + bs[0] + maskp[t * B_DIM + b];
    }
}

// ---------------------------------------------------------------------------
// Kernel 4: softmax over the T dimension, one block per b.
// ---------------------------------------------------------------------------
__global__ void softmax_t_kernel(const float* __restrict__ scores,
                                 float* __restrict__ out) {
    __shared__ float red[256];
    const int b   = blockIdx.x;
    const int tid = threadIdx.x;

    float m = -1e30f;
    for (int t = tid; t < T_DIM; t += 256)
        m = fmaxf(m, scores[t * B_DIM + b]);
    red[tid] = m; __syncthreads();
    for (int s = 128; s > 0; s >>= 1) {
        if (tid < s) red[tid] = fmaxf(red[tid], red[tid + s]);
        __syncthreads();
    }
    const float mx = red[0]; __syncthreads();

    float sum = 0.0f;
    for (int t = tid; t < T_DIM; t += 256)
        sum += expf(scores[t * B_DIM + b] - mx);
    red[tid] = sum; __syncthreads();
    for (int s = 128; s > 0; s >>= 1) {
        if (tid < s) red[tid] += red[tid + s];
        __syncthreads();
    }
    const float inv = 1.0f / red[0];

    for (int t = tid; t < T_DIM; t += 256)
        out[t * B_DIM + b] = expf(scores[t * B_DIM + b] - mx) * inv;
}

// ---------------------------------------------------------------------------
extern "C" void kernel_launch(void* const* d_in, const int* in_sizes, int n_in,
                              void* d_out, int out_size, void* d_ws, size_t ws_size,
                              hipStream_t stream) {
    const float* enc  = (const float*)d_in[0];  // (T,B,H)
    const float* mask = (const float*)d_in[1];  // (T,B)
    const float* lm   = (const float*)d_in[2];  // (B,L)
    const float* prev = (const float*)d_in[3];  // (T,B)
    const float* Wk   = (const float*)d_in[4];  // (K*H, L)
    const float* bk   = (const float*)d_in[5];  // (K*H,)
    const float* Wg   = (const float*)d_in[6];  // (H, L)
    const float* bg   = (const float*)d_in[7];  // (H,)
    const float* Ws   = (const float*)d_in[8];  // (1, H)
    const float* bs   = (const float*)d_in[9];  // (1,)

    char* ws = (char*)d_ws;
    _Float16* kern = (_Float16*)ws;                       // 32*8192 f16 = 512 KB
    float* ghid    = (float*)(ws + 524288);               // 32*256 f32  =  32 KB
    float* scores  = (float*)(ws + 524288 + 32768);       // 4096*32 f32 = 512 KB
    float* out     = (float*)d_out;                       // (T,B)

    wk_gemm_kernel<<<128, 256, 0, stream>>>(lm, Wk, bk, kern);
    ghid_kernel<<<32, 256, 0, stream>>>(lm, Wg, bg, ghid);
    fused_scores_kernel<<<1024, 256, 0, stream>>>(enc, mask, prev, kern, ghid,
                                                  Ws, bs, scores);
    softmax_t_kernel<<<32, 256, 0, stream>>>(scores, out);
}
